// GraphAdvection_49460843380759
// MI455X (gfx1250) — matrix operand order repaired and checked by reading, not compiled, not run on hardware
//
#include <hip/hip_runtime.h>
#include <hip/hip_bf16.h>

typedef __attribute__((ext_vector_type(16))) __bf16 v16bf;
typedef __attribute__((ext_vector_type(8)))  __bf16 bf16x8;
typedef __attribute__((ext_vector_type(4)))  __bf16 bf16x4;
typedef __attribute__((ext_vector_type(8)))  float  v8f;
typedef __attribute__((ext_vector_type(4)))  float  v4f;
typedef __attribute__((ext_vector_type(4)))  unsigned u32x4;

#define B_    4
#define N_    1000
#define T_    12
#define D_    64
#define HID_  128
#define EF_   64
#define E_    64000
#define STEP_ 0.1f
#define NODES (B_*N_)        /* 4000 */
#define ROWS  (NODES*T_)     /* 48000 rows of the flattened [B*N*T, D] x */
#define ND    (T_*D_)        /* 768 elements per node */
#define NELEM (NODES*ND)     /* 3,072,000 */

// ---------------- helpers ----------------

__device__ __forceinline__ unsigned encF(float f) {
    unsigned u = __float_as_uint(f);
    return (u & 0x80000000u) ? ~u : (u | 0x80000000u);
}
__device__ __forceinline__ float decF(unsigned k) {
    unsigned u = (k & 0x80000000u) ? (k & 0x7FFFFFFFu) : ~k;
    return __uint_as_float(u);
}

__device__ __forceinline__ v8f wmma_bf16(v16bf a, v16bf b, v8f c) {
    // D = A(16x32 bf16) * B(32x16 bf16) + C(16x16 f32)
    return __builtin_amdgcn_wmma_f32_16x16x32_bf16(
        /*neg_a=*/false, a, /*neg_b=*/false, b,
        /*c_mod=*/(short)0, c, /*reuse_a=*/false, /*reuse_b=*/false);
}

__device__ __forceinline__ v16bf cat8(bf16x8 lo, bf16x8 hi) {
    return __builtin_shufflevector(lo, hi, 0, 1, 2, 3, 4, 5, 6, 7,
                                   8, 9, 10, 11, 12, 13, 14, 15);
}

// A-fragment from a row-major bf16 row pointer (already offset by kb):
// slots 0..7 = K(kb)+8*kh .. +7 ; slots 8..15 = K(kb)+16+8*kh .. +7
__device__ __forceinline__ v16bf loadAfrag(const __bf16* rowkb, int kh) {
    bf16x8 lo = *(const bf16x8*)(rowkb + 8 * kh);
    bf16x8 hi = *(const bf16x8*)(rowkb + 16 + 8 * kh);
    return cat8(lo, hi);
}

// B-fragment from pre-packed weights: one aligned 32-byte load (2x b128).
// Packed layout: [nt][kb32][lane][16 slots] contiguous.
__device__ __forceinline__ v16bf loadBfrag(const __bf16* Wp, int nt, int nKb,
                                           int kb32, int lane) {
    return *(const v16bf*)(Wp + ((((size_t)nt * nKb + kb32) * 32 + lane) << 4));
}

// ---------------- init kernels ----------------

__global__ void kInit(const float* __restrict__ x, v4f* __restrict__ denom4,
                      v4f* __restrict__ inbound4, u32x4* __restrict__ segmax4,
                      bf16x4* __restrict__ xbf4) {
    int i = blockIdx.x * blockDim.x + threadIdx.x;   // over NELEM/4
    if (i < NELEM / 4) {
        v4f z = {};
        u32x4 zu = {};
        denom4[i]   = z;
        inbound4[i] = z;
        segmax4[i]  = zu;            // decodes below any real value
        v4f xv = *(const v4f*)(x + 4 * (size_t)i);
        bf16x4 o;
        o.x = (__bf16)xv.x; o.y = (__bf16)xv.y;
        o.z = (__bf16)xv.z; o.w = (__bf16)xv.w;
        xbf4[i] = o;
    }
}

// Pack weight matrix W[K][Ncols] (f32) into B-fragment order (bf16):
// P[(((nt*(K/32) + kb32)*32) + lane)*16 + slot] = W[k][n]
//   k = kb32*32 + 8*(lane>>4) + (slot&7) + 16*(slot>>3) ; n = nt*16 + (lane&15)
__device__ __forceinline__ void packW(const float* __restrict__ W,
                                      __bf16* __restrict__ P,
                                      int i, int K, int Ncols) {
    int slot = i & 15;
    int lane = (i >> 4) & 31;
    int blk  = i >> 9;
    int nKb  = K >> 5;
    int kb32 = blk % nKb;
    int nt   = blk / nKb;
    int k = kb32 * 32 + 8 * (lane >> 4) + (slot & 7) + 16 * (slot >> 3);
    int n = nt * 16 + (lane & 15);
    P[i] = (__bf16)W[k * Ncols + n];
}

__global__ void kInitW(const float* __restrict__ w1, const float* __restrict__ w2,
                       const float* __restrict__ w3, const float* __restrict__ w4,
                       const float* __restrict__ b1, const float* __restrict__ b2,
                       __bf16* __restrict__ w1p, __bf16* __restrict__ w2p,
                       __bf16* __restrict__ w3p, __bf16* __restrict__ w4p,
                       float* __restrict__ bsum) {
    int i = blockIdx.x * blockDim.x + threadIdx.x;
    if (i < D_ * HID_) {                 // 8192 == HID_*EF_ too
        packW(w1, w1p, i, D_, HID_);     // A1_w: K=64,  N=128
        packW(w2, w2p, i, D_, HID_);     // A2_w: K=64,  N=128
        packW(w3, w3p, i, HID_, EF_);    // A3_w: K=128, N=64
    }
    if (i < EF_ * D_) packW(w4, w4p, i, EF_, D_);  // A4_w: K=64, N=64
    if (i < HID_)     bsum[i] = b1[i] + b2[i];
}

// ---------------- node projection: U = x@A1, V = x@A2 ----------------
// One wave per 16x16 m/n tile computes BOTH U and V tiles (shared A-frags):
// K=64 -> 2 WMMAs per output matrix, 4 per wave.

__global__ void __launch_bounds__(256) kNodeProj(
    const __bf16* __restrict__ xbf, const __bf16* __restrict__ w1p,
    const __bf16* __restrict__ w2p, float* __restrict__ U, float* __restrict__ V) {
    int lane = threadIdx.x & 31;
    int gw   = blockIdx.x * (blockDim.x >> 5) + (threadIdx.x >> 5);
    int mt = gw >> 3;          // 0..2999
    int nt = gw & 7;           // column tile of 128

    int m  = lane & 15;
    int kh = lane >> 4;
    int row = mt * 16 + m;
    int col = nt * 16 + m;
    const __bf16* arow = xbf + (size_t)row * D_;

    v8f accU = {}, accV = {};
    #pragma unroll
    for (int kb = 0; kb < D_; kb += 32) {
        v16bf a  = loadAfrag(arow + kb, kh);
        v16bf b0 = loadBfrag(w1p, nt, D_ >> 5, kb >> 5, lane);
        v16bf b1 = loadBfrag(w2p, nt, D_ >> 5, kb >> 5, lane);
        accU = wmma_bf16(a, b0, accU);
        accV = wmma_bf16(a, b1, accV);
    }
    #pragma unroll
    for (int r = 0; r < 8; r++) {
        size_t o = (size_t)(mt * 16 + r + 8 * kh) * HID_ + col;
        U[o] = accU[r];
        V[o] = accV[r];
    }
}

// ---------------- edge kernel ----------------
// Block = 128 threads (4 waves) per edge. d = h_ij - h_ji in fp32 -> bf16 LDS.
// GEMM1: [16x128]@[128x64] (4 WMMAs/wave), relu -> LDS.
// GEMM2: [16x64]@[64x64] (2 WMMAs/wave), +b4, store + segment atomicMax.

__global__ void __launch_bounds__(128) kEdge(
    const int* __restrict__ bi, const int* __restrict__ si, const int* __restrict__ ri,
    const float* __restrict__ U, const float* __restrict__ V,
    const float* __restrict__ bsum, const __bf16* __restrict__ w3p,
    const __bf16* __restrict__ w4p, const float* __restrict__ b4,
    float* __restrict__ vout, unsigned* __restrict__ segmax) {
    __shared__ __align__(32) __bf16 dsh[16 * HID_];
    __shared__ __align__(32) __bf16 gsh[16 * EF_];

    int e   = blockIdx.x;
    int tid = threadIdx.x;
    int b = bi[e], s = si[e], r = ri[e];
    int ns = b * N_ + s;
    int nr = b * N_ + r;
    const float* Us = U + (size_t)ns * T_ * HID_;
    const float* Vr = V + (size_t)nr * T_ * HID_;
    const float* Ur = U + (size_t)nr * T_ * HID_;
    const float* Vs = V + (size_t)ns * T_ * HID_;

    // d[t][c..c+3] = relu(Us+Vr+b) - relu(Ur+Vs+b); rows 12..15 padded zero.
    // 512 four-wide chunks / 128 threads = 4 iterations, all b128 loads.
    for (int it = tid; it < 512; it += 128) {
        int t = it >> 5;             // 0..15
        int c = (it & 31) << 2;      // 0,4,..,124
        v4f dv = {};
        if (t < T_) {
            v4f bs = *(const v4f*)(bsum + c);
            v4f us = *(const v4f*)(Us + t * HID_ + c);
            v4f vr = *(const v4f*)(Vr + t * HID_ + c);
            v4f ur = *(const v4f*)(Ur + t * HID_ + c);
            v4f vs = *(const v4f*)(Vs + t * HID_ + c);
            v4f hij = us + vr + bs;
            v4f hji = ur + vs + bs;
            dv.x = fmaxf(hij.x, 0.f) - fmaxf(hji.x, 0.f);
            dv.y = fmaxf(hij.y, 0.f) - fmaxf(hji.y, 0.f);
            dv.z = fmaxf(hij.z, 0.f) - fmaxf(hji.z, 0.f);
            dv.w = fmaxf(hij.w, 0.f) - fmaxf(hji.w, 0.f);
        }
        bf16x4 o;
        o.x = (__bf16)dv.x; o.y = (__bf16)dv.y;
        o.z = (__bf16)dv.z; o.w = (__bf16)dv.w;
        *(bf16x4*)(dsh + t * HID_ + c) = o;
    }
    __syncthreads();

    int w    = tid >> 5;
    int lane = tid & 31;
    int m    = lane & 15;
    int kh   = lane >> 4;
    int ncol = w * 16 + m;
    const __bf16* drow = dsh + m * HID_;

    v8f acc = {};
    #pragma unroll
    for (int kb = 0; kb < HID_; kb += 32) {
        v16bf a  = loadAfrag(drow + kb, kh);
        v16bf bm = loadBfrag(w3p, w, HID_ >> 5, kb >> 5, lane);
        acc = wmma_bf16(a, bm, acc);
    }
    #pragma unroll
    for (int rr = 0; rr < 8; rr++) {
        int orow = rr + 8 * kh;
        gsh[orow * EF_ + ncol] = (__bf16)fmaxf(acc[rr], 0.f);
    }
    __syncthreads();

    const __bf16* grow = gsh + m * EF_;
    v8f acc2 = {};
    #pragma unroll
    for (int kb = 0; kb < EF_; kb += 32) {
        v16bf a  = loadAfrag(grow + kb, kh);
        v16bf bm = loadBfrag(w4p, w, EF_ >> 5, kb >> 5, lane);
        acc2 = wmma_bf16(a, bm, acc2);
    }

    float bias = b4[ncol];
    #pragma unroll
    for (int rr = 0; rr < 8; rr++) {
        int orow = rr + 8 * kh;
        if (orow < T_) {
            float val = acc2[rr] + bias;
            vout[(size_t)e * ND + orow * D_ + ncol] = val;
            atomicMax(&segmax[ns * ND + orow * D_ + ncol], encF(val));
        }
    }
}

// ---------------- softmax passes + scatter + update ----------------
// One block (192 threads) per edge; each thread owns 4 consecutive elements.
// Edge index is uniform -> batch/sender/receiver lookups become scalar loads.

__global__ void __launch_bounds__(192) kExpDenom(
    const int* __restrict__ bi, const int* __restrict__ si,
    float* __restrict__ vtmp, const unsigned* __restrict__ segmax,
    float* __restrict__ denom) {
    int e   = blockIdx.x;
    int rem = threadIdx.x << 2;          // 0..764
    int ns  = bi[e] * N_ + si[e];
    size_t i = (size_t)e * ND + rem;
    int sb = ns * ND + rem;

    v4f p = *(const v4f*)(vtmp + i);
    u32x4 mk = *(const u32x4*)(segmax + sb);
    v4f ex;
    ex.x = __expf(p.x - decF(mk.x));
    ex.y = __expf(p.y - decF(mk.y));
    ex.z = __expf(p.z - decF(mk.z));
    ex.w = __expf(p.w - decF(mk.w));
    *(v4f*)(vtmp + i) = ex;
    atomicAdd(&denom[sb + 0], ex.x);
    atomicAdd(&denom[sb + 1], ex.y);
    atomicAdd(&denom[sb + 2], ex.z);
    atomicAdd(&denom[sb + 3], ex.w);
}

__global__ void __launch_bounds__(192) kNorm(
    const int* __restrict__ bi, const int* __restrict__ si,
    const int* __restrict__ ri, const float* __restrict__ x,
    float* __restrict__ vtmp, const float* __restrict__ denom,
    float* __restrict__ inbound) {
    int e   = blockIdx.x;
    int rem = threadIdx.x << 2;
    int b   = bi[e];
    int ns  = b * N_ + si[e];
    int nr  = b * N_ + ri[e];
    size_t i = (size_t)e * ND + rem;

    v4f ex = *(const v4f*)(vtmp + i);
    v4f dn = *(const v4f*)(denom + ns * ND + rem);
    v4f xs = *(const v4f*)(x + (size_t)ns * ND + rem);
    v4f v;
    v.x = ex.x / (dn.x + 1e-12f);
    v.y = ex.y / (dn.y + 1e-12f);
    v.z = ex.z / (dn.z + 1e-12f);
    v.w = ex.w / (dn.w + 1e-12f);
    *(v4f*)(vtmp + i) = v;               // final v_ij output
    int ob = nr * ND + rem;
    atomicAdd(&inbound[ob + 0], v.x * xs.x);
    atomicAdd(&inbound[ob + 1], v.y * xs.y);
    atomicAdd(&inbound[ob + 2], v.z * xs.z);
    atomicAdd(&inbound[ob + 3], v.w * xs.w);
}

__global__ void kUpdate(const float* __restrict__ x, const float* __restrict__ inbound,
                        float* __restrict__ out) {
    int i = blockIdx.x * blockDim.x + threadIdx.x;   // over NELEM/4
    if (i < NELEM / 4) {
        v4f xv = *(const v4f*)(x + 4 * (size_t)i);
        v4f ib = *(const v4f*)(inbound + 4 * (size_t)i);
        v4f o  = xv + (ib - xv) * STEP_;
        *(v4f*)((float*)out + 4 * (size_t)i) = o;
    }
}

// ---------------- launch ----------------

extern "C" void kernel_launch(void* const* d_in, const int* in_sizes, int n_in,
                              void* d_out, int out_size, void* d_ws, size_t ws_size,
                              hipStream_t stream) {
    const float* x  = (const float*)d_in[0];
    const int*   bi = (const int*)d_in[1];
    const int*   si = (const int*)d_in[2];
    const int*   ri = (const int*)d_in[3];
    const float* w1 = (const float*)d_in[4];
    const float* b1 = (const float*)d_in[5];
    const float* w2 = (const float*)d_in[6];
    const float* b2 = (const float*)d_in[7];
    const float* w3 = (const float*)d_in[8];
    // d_in[9] = A3_b: cancels in z_ij - z_ji, unused
    const float* w4 = (const float*)d_in[10];
    const float* b4 = (const float*)d_in[11];

    float* out  = (float*)d_out;      // [B,N,T,D] x_updated
    float* vout = out + NELEM;        // [E,T,D]   v_ij (used as scratch in passes)

    // workspace carve-up (~92 MB total); all sub-buffers stay 32B-aligned
    float*    U       = (float*)d_ws;                      // [ROWS, HID]
    float*    V       = U + (size_t)ROWS * HID_;           // [ROWS, HID]
    float*    denom   = V + (size_t)ROWS * HID_;           // [NELEM]
    float*    inbound = denom + NELEM;                     // [NELEM]
    unsigned* segmax  = (unsigned*)(inbound + NELEM);      // [NELEM]
    __bf16*   xbf     = (__bf16*)(segmax + NELEM);         // [NELEM]
    __bf16*   w1p     = xbf + NELEM;                       // packed [D*HID]
    __bf16*   w2p     = w1p + D_ * HID_;
    __bf16*   w3p     = w2p + D_ * HID_;                   // packed [HID*EF]
    __bf16*   w4p     = w3p + HID_ * EF_;                  // packed [EF*D]
    float*    bsum    = (float*)(w4p + EF_ * D_);          // [HID]

    kInit<<<(NELEM / 4 + 255) / 256, 256, 0, stream>>>(
        x, (v4f*)denom, (v4f*)inbound, (u32x4*)segmax, (bf16x4*)xbf);
    kInitW<<<(D_ * HID_ + 255) / 256, 256, 0, stream>>>(w1, w2, w3, w4, b1, b2,
                                                        w1p, w2p, w3p, w4p, bsum);
    // 3000 m-tiles * 8 n-tiles = 24000 waves / 8 waves-per-block
    kNodeProj<<<3000, 256, 0, stream>>>(xbf, w1p, w2p, U, V);
    kEdge<<<E_, 128, 0, stream>>>(bi, si, ri, U, V, bsum, w3p, w4p, b4, vout, segmax);
    kExpDenom<<<E_, 192, 0, stream>>>(bi, si, vout, segmax, denom);
    kNorm<<<E_, 192, 0, stream>>>(bi, si, ri, x, vout, denom, inbound);
    kUpdate<<<(NELEM / 4 + 255) / 256, 256, 0, stream>>>(x, inbound, out);
}
